// MultiHeadSelfAttention_5531917877641
// MI455X (gfx1250) — matrix operand order repaired and checked by reading, not compiled, not run on hardware
//
#include <hip/hip_runtime.h>

#define B_  4
#define S_  2048
#define D_  1024
#define H_  16
#define HD_ 64

typedef __attribute__((ext_vector_type(16))) __bf16 bfx16;
typedef __attribute__((ext_vector_type(8)))  __bf16 bfx8;
typedef __attribute__((ext_vector_type(4)))  __bf16 bfx4;
typedef __attribute__((ext_vector_type(8)))  float  fx8;

union BF16x16 { bfx16 v; bfx8 h[2]; };

__device__ __forceinline__ fx8 wmma_bf16(bfx16 a, bfx16 b, fx8 c) {
  return __builtin_amdgcn_wmma_f32_16x16x32_bf16(false, a, false, b, (short)0, c, false, false);
}

__device__ __forceinline__ float red_max16(float v) {
  v = fmaxf(v, __shfl_xor(v, 1));
  v = fmaxf(v, __shfl_xor(v, 2));
  v = fmaxf(v, __shfl_xor(v, 4));
  v = fmaxf(v, __shfl_xor(v, 8));
  return v;
}
__device__ __forceinline__ float red_sum16(float v) {
  v += __shfl_xor(v, 1);
  v += __shfl_xor(v, 2);
  v += __shfl_xor(v, 4);
  v += __shfl_xor(v, 8);
  return v;
}

// ---------------- conversion kernels ----------------

__global__ void __launch_bounds__(256) cvt_x_kernel(const float* __restrict__ x,
                                                    __bf16* __restrict__ xb) {
  int i = blockIdx.x * 256 + threadIdx.x;          // n/4 threads
  const float4 f = ((const float4*)x)[i];
  bfx4 o;
  o[0] = (__bf16)f.x; o[1] = (__bf16)f.y; o[2] = (__bf16)f.z; o[3] = (__bf16)f.w;
  ((bfx4*)xb)[i] = o;
}

// W [D, 3D] f32  ->  Wt [3D, D] bf16 (transposed so GEMM B-frags are contiguous)
__global__ void __launch_bounds__(256) cvt_w_kernel(const float* __restrict__ W,
                                                    __bf16* __restrict__ Wt) {
  int t = blockIdx.x * 256 + threadIdx.x;          // 3D*D threads
  int d = t & (D_ - 1);
  int n = t >> 10;
  Wt[(size_t)n * D_ + d] = (__bf16)W[(size_t)d * (3 * D_) + n];
}

// ---------------- fused QKV GEMM (bf16 WMMA) ----------------
// Each wave: 16x64 output tile. M = B*S = 8192, K = D = 1024, N = 3*D = 3072.
// Q, K stored [B,H,S,64] (Q pre-scaled by 1/sqrt(64)); V stored transposed
// [B,H,64,S] so attention's P@V B-fragments are contiguous global loads.
__global__ void __launch_bounds__(256) qkv_gemm_kernel(
    const __bf16* __restrict__ xb, const __bf16* __restrict__ wt,
    const float*  __restrict__ bias,
    __bf16* __restrict__ qb, __bf16* __restrict__ kb, __bf16* __restrict__ vb)
{
  const int lane = threadIdx.x & 31;
  const int wid  = blockIdx.x * 8 + (threadIdx.x >> 5);
  const int nTiles = (3 * D_) / 64;                // 48
  const int mTile  = wid / nTiles;                 // 0..511
  const int nTile  = wid % nTiles;
  const int row0 = mTile * 16;
  const int n0   = nTile * 64;

  const int mA    = lane & 15;
  const int koffA = (lane >> 4) * 8;               // A frag K sub-offset
  const int nB    = lane & 15;
  const int kbB   = (lane >> 4) * 8;               // B frag K base
  const int mbase = (lane >> 4) * 8;               // C frag row base

  fx8 c[4] = {};

  const __bf16* aRow = xb + (size_t)(row0 + mA) * D_;
  for (int k0 = 0; k0 < D_; k0 += 32) {
    BF16x16 a;
    a.h[0] = *(const bfx8*)(aRow + k0 + koffA);
    a.h[1] = *(const bfx8*)(aRow + k0 + 16 + koffA);
#pragma unroll
    for (int f = 0; f < 4; ++f) {
      const __bf16* bp = wt + (size_t)(n0 + f * 16 + nB) * D_ + k0 + kbB;
      BF16x16 b;
      b.h[0] = *(const bfx8*)(bp);
      b.h[1] = *(const bfx8*)(bp + 16);
      c[f] = wmma_bf16(a.v, b.v, c[f]);
    }
  }

  // epilogue: bias; scatter into per-head buffers
  const int h  = n0 / 192;
  const int r0 = n0 % 192;                          // 0 -> Q, 64 -> K, 128 -> V
  const int row  = row0 + mbase;                    // first of this lane's 8 rows
  const int bb   = row >> 11;                       // / S_ (8 rows never cross b)
  const int s0   = row & (S_ - 1);

  if (r0 == 128) {
    // V: transposed store [B,H,64,S]; 8 rows are contiguous in s -> one b128
#pragma unroll
    for (int f = 0; f < 4; ++f) {
      const int col = f * 16 + (lane & 15);
      const float bv = bias[n0 + col];
      bfx8 pk;
#pragma unroll
      for (int r = 0; r < 8; ++r) pk[r] = (__bf16)(c[f][r] + bv);
      *(bfx8*)(vb + ((size_t)(bb * H_ + h) * HD_ + col) * S_ + s0) = pk;
    }
  } else {
    __bf16* dst = (r0 == 0) ? qb : kb;
    const float qscale = (r0 == 0) ? 0.125f : 1.0f;   // fold 1/sqrt(hd) into Q
#pragma unroll
    for (int f = 0; f < 4; ++f) {
      const int col = f * 16 + (lane & 15);
      const float bv = bias[n0 + col];
#pragma unroll
      for (int r = 0; r < 8; ++r) {
        const float val = (c[f][r] + bv) * qscale;
        dst[(size_t)((bb * H_ + h) * S_ + (s0 + r)) * HD_ + col] = (__bf16)val;
      }
    }
  }
}

// ---------------- causal flash attention (bf16 WMMA) ----------------
#define PSTRIDE  32
#define WAVE_LDS (16 * PSTRIDE)                      // 512 bf16 per wave (P staging)

__global__ void __launch_bounds__(256) attn_kernel(
    const __bf16* __restrict__ qb, const __bf16* __restrict__ kb,
    const __bf16* __restrict__ vb, float* __restrict__ out)
{
  __shared__ __bf16 lds[8 * WAVE_LDS];               // 8 KB
  const int lane = threadIdx.x & 31;
  const int warp = threadIdx.x >> 5;
  const int wid  = blockIdx.x * 8 + warp;            // 0..8191
  const int qt   = wid & 127;                        // S/16 tiles
  const int bh   = wid >> 7;                         // 0..63
  const int b    = bh >> 4;
  const int h    = bh & 15;
  const int q0   = qt * 16;

  __bf16* pbuf = lds + warp * WAVE_LDS;

  const __bf16* Q  = qb + (size_t)(b * H_ + h) * S_ * HD_;
  const __bf16* K  = kb + (size_t)(b * H_ + h) * S_ * HD_;
  const __bf16* Vt = vb + (size_t)(b * H_ + h) * HD_ * S_;   // [64][S]

  const int mA    = lane & 15;
  const int koffA = (lane >> 4) * 8;
  const int nloc  = lane & 15;
  const int kbB   = (lane >> 4) * 8;
  const int mbase = (lane >> 4) * 8;

  // Q A-fragments (16 rows x 64, pre-scaled by 1/sqrt(hd) at GEMM store)
  BF16x16 aq0, aq1;
  {
    const __bf16* qr = Q + (size_t)(q0 + mA) * HD_;
    aq0.h[0] = *(const bfx8*)(qr + koffA);
    aq0.h[1] = *(const bfx8*)(qr + 16 + koffA);
    aq1.h[0] = *(const bfx8*)(qr + 32 + koffA);
    aq1.h[1] = *(const bfx8*)(qr + 48 + koffA);
  }

  fx8 o[4] = {};
  float rm[8], rl[8];
#pragma unroll
  for (int r = 0; r < 8; ++r) { rm[r] = -1e30f; rl[r] = 0.f; }

  for (int j0 = 0; j0 < q0 + 16; j0 += 32) {
    // ---- S tile: Q @ K^T for keys [j0, j0+32) ----
    fx8 s1 = {}, s2 = {};
    {
      const __bf16* kr1 = K + (size_t)(j0 + nloc) * HD_ + kbB;
      const __bf16* kr2 = K + (size_t)(j0 + 16 + nloc) * HD_ + kbB;
      BF16x16 bk;
      bk.h[0] = *(const bfx8*)(kr1);      bk.h[1] = *(const bfx8*)(kr1 + 16);
      s1 = wmma_bf16(aq0.v, bk.v, s1);
      bk.h[0] = *(const bfx8*)(kr1 + 32); bk.h[1] = *(const bfx8*)(kr1 + 48);
      s1 = wmma_bf16(aq1.v, bk.v, s1);
      bk.h[0] = *(const bfx8*)(kr2);      bk.h[1] = *(const bfx8*)(kr2 + 16);
      s2 = wmma_bf16(aq0.v, bk.v, s2);
      bk.h[0] = *(const bfx8*)(kr2 + 32); bk.h[1] = *(const bfx8*)(kr2 + 48);
      s2 = wmma_bf16(aq1.v, bk.v, s2);
    }

    // prefetch next key block of K / Vt toward the WGP while softmax runs
    if (j0 + 32 < q0 + 16) {
      __builtin_prefetch((const void*)(K + (size_t)(j0 + 32 + nloc) * HD_ + kbB), 0, 0);
      __builtin_prefetch((const void*)(Vt + (size_t)nloc * S_ + j0 + 32 + kbB), 0, 0);
    }

    // ---- causal mask (diagonal blocks only) ----
    if (j0 + 31 > q0) {
#pragma unroll
      for (int r = 0; r < 8; ++r) {
        const int qrow = q0 + mbase + r;
        if (j0 + nloc > qrow)      s1[r] = -1e30f;
        if (j0 + 16 + nloc > qrow) s2[r] = -1e30f;
      }
    }

    // ---- online softmax; stage P (C-layout -> LDS) ----
    float scale[8];
#pragma unroll
    for (int r = 0; r < 8; ++r) {
      float mx = red_max16(fmaxf(s1[r], s2[r]));
      float mn = fmaxf(rm[r], mx);
      float p1 = __expf(s1[r] - mn);
      float p2 = __expf(s2[r] - mn);
      float sum = red_sum16(p1 + p2);
      float sc  = __expf(rm[r] - mn);
      rl[r] = rl[r] * sc + sum;
      rm[r] = mn;
      scale[r] = sc;
      pbuf[(mbase + r) * PSTRIDE + nloc]      = (__bf16)p1;
      pbuf[(mbase + r) * PSTRIDE + 16 + nloc] = (__bf16)p2;
    }
#pragma unroll
    for (int f = 0; f < 4; ++f)
#pragma unroll
      for (int r = 0; r < 8; ++r)
        o[f][r] *= scale[r];

    // wave-local LDS fence (waves in this block run divergent trip counts,
    // each has a private LDS slice, so no block barrier — just drain DScnt)
    asm volatile("s_wait_dscnt 0" ::: "memory");

    // ---- P as A-fragment (16 x 32) from LDS ----
    BF16x16 ap;
    ap.h[0] = *(const bfx8*)(pbuf + mA * PSTRIDE + koffA);
    ap.h[1] = *(const bfx8*)(pbuf + mA * PSTRIDE + 16 + koffA);

    // ---- O += P @ V : B-frags are contiguous rows of Vt ----
#pragma unroll
    for (int f = 0; f < 4; ++f) {
      const __bf16* vp = Vt + (size_t)(f * 16 + nloc) * S_ + j0 + kbB;
      BF16x16 bvf;
      bvf.h[0] = *(const bfx8*)(vp);
      bvf.h[1] = *(const bfx8*)(vp + 16);
      o[f] = wmma_bf16(ap.v, bvf.v, o[f]);
    }
  }

  // ---- normalize and store f32 [B, S, D] ----
#pragma unroll
  for (int f = 0; f < 4; ++f) {
    const int col = h * HD_ + f * 16 + nloc;
#pragma unroll
    for (int r = 0; r < 8; ++r) {
      const int srow = q0 + mbase + r;
      out[(size_t)(b * S_ + srow) * D_ + col] = o[f][r] / rl[r];
    }
  }
}

// ---------------- host launcher ----------------
extern "C" void kernel_launch(void* const* d_in, const int* in_sizes, int n_in,
                              void* d_out, int out_size, void* d_ws, size_t ws_size,
                              hipStream_t stream) {
  (void)in_sizes; (void)n_in; (void)out_size; (void)ws_size;
  const float* x    = (const float*)d_in[0];
  const float* W    = (const float*)d_in[1];
  const float* bias = (const float*)d_in[2];
  float* out = (float*)d_out;

  char* ws = (char*)d_ws;
  const size_t nX   = (size_t)B_ * S_ * D_;       // 8,388,608
  const size_t nW   = (size_t)3 * D_ * D_;        // 3,145,728
  const size_t nQKV = (size_t)B_ * H_ * S_ * HD_; // 8,388,608 each
  __bf16* xb = (__bf16*)ws;
  __bf16* wt = (__bf16*)(ws + nX * 2);
  __bf16* qb = (__bf16*)(ws + (nX + nW) * 2);
  __bf16* kb = (__bf16*)(ws + (nX + nW + nQKV) * 2);
  __bf16* vb = (__bf16*)(ws + (nX + nW + 2 * nQKV) * 2);

  // 1) convert x -> bf16
  cvt_x_kernel<<<(int)(nX / 4 / 256), 256, 0, stream>>>(x, xb);
  // 2) convert + transpose W -> bf16 [3D, D]
  cvt_w_kernel<<<(int)(nW / 256), 256, 0, stream>>>(W, wt);
  // 3) fused QKV projection -> bf16 Q/K [B,H,S,64], V [B,H,64,S]
  //    waves = (8192/16) * (3072/64) = 24576; 8 waves/block
  qkv_gemm_kernel<<<24576 / 8, 256, 0, stream>>>(xb, wt, bias, qb, kb, vb);
  // 4) causal flash attention -> f32 out [B,S,D]
  //    waves = B*H*(S/16) = 8192; 8 waves/block
  attn_kernel<<<8192 / 8, 256, 0, stream>>>(qb, kb, vb, out);
}
